// ScOTLayer_69973607186742
// MI455X (gfx1250) — matrix operand order repaired and checked by reading, not compiled, not run on hardware
//
#include <hip/hip_runtime.h>
#include <cstdint>
#include <cstddef>

// ---------------------------------------------------------------------------
// Problem constants (from reference): B=16, H=W=64, C=192, WS=16, SHIFT=8,
// NH=6, D=32, N=256 tokens/window, 256 windows total, HID=768.
// ---------------------------------------------------------------------------
#define TOKENS   65536        // B*H*W
#define CH       192
#define NHEAD    6
#define HDIM     32
#define WINN     256          // tokens per window
#define NWINTOT  256          // B * nW
#define HIDCH    768
#define LN_EPS   1e-5f

// Async global->LDS (CDNA5): use if the toolchain exposes the builtin.
#if defined(__has_builtin)
#if __has_builtin(__builtin_amdgcn_global_load_async_to_lds_b128)
#define USE_ASYNC_LDS 1
#endif
#endif

// ---------------------------------------------------------------------------
// bf16 WMMA plumbing (gfx1250: V_WMMA_F32_16X16X32_BF16, wave32)
// ---------------------------------------------------------------------------
typedef __bf16 bf16_t;
typedef bf16_t v16bf __attribute__((ext_vector_type(16)));
typedef float  v8f   __attribute__((ext_vector_type(8)));
typedef int    v4i   __attribute__((vector_size(16)));   // int4 for async-LDS builtin

#if USE_ASYNC_LDS
typedef __attribute__((address_space(1))) v4i* gv4i_p;   // global int4*
typedef __attribute__((address_space(3))) v4i* lv4i_p;   // LDS int4*
#endif

union FragU { v16bf v; uint32_t u[8]; };

__device__ __forceinline__ uint16_t f2bf(float f) {
  uint32_t u = __float_as_uint(f);
  u += 0x7fffu + ((u >> 16) & 1u);          // round-to-nearest-even
  return (uint16_t)(u >> 16);
}

__device__ __forceinline__ v8f wmma_bf16(const FragU& a, const FragU& b, v8f c) {
  return __builtin_amdgcn_wmma_f32_16x16x32_bf16(false, a.v, false, b.v,
                                                 (short)0, c, false, false);
}

// A-fragment K index for 16-bit A 16x32 (contiguous order, matches B): pairs at
// lanes 0-15: v0..7 -> K 0..15; lanes 16-31: +16.
__device__ __forceinline__ int kkA(int v, int half) {
  return 2 * v + half * 16;
}

// ---------------------------------------------------------------------------
// Workspace layout (bytes). Regions are reused only after producers are dead.
// Total required: ~258.9 MB.
// ---------------------------------------------------------------------------
static const size_t OFF_A = 0;              // xw bf16 (25165824) ; later h bf16
static const size_t OFF_B = 25165824;       // q f32 ; later attn-out bf16 ; later mlp bf16 (spans B+C)
static const size_t OFF_C = 75497472;       // k f32 ; later proj f32
static const size_t OFF_D = 125829120;      // v f32 ; later h f32
static const size_t OFF_E = 176160768;      // qn bf16 ; later fc2 f32 (spans E+F)
static const size_t OFF_F = 201326592;      // kn bf16
static const size_t OFF_G = 226492416;      // vT bf16 (25165824)
static const size_t OFF_H = 251658240;      // cpb bias table 961*6*4
static const size_t OFF_W = 251691008;      // bf16 weights (884736)
static const size_t OFF_I = 252575744;      // bias+mask table 4*6*256*256*4

// weight element offsets inside OFF_W
#define WOFF_Q  0
#define WOFF_K  36864
#define WOFF_V  73728
#define WOFF_P  110592
#define WOFF_F1 147456
#define WOFF_F2 294912

// ---------------------------------------------------------------------------
// fp32 -> bf16 convert (weights)
// ---------------------------------------------------------------------------
__global__ __launch_bounds__(256) void k_cvt(const float* __restrict__ s,
                                             uint16_t* __restrict__ d, int n) {
  int i = blockIdx.x * 256 + threadIdx.x;
  if (i < n) d[i] = f2bf(s[i]);
}

// ---------------------------------------------------------------------------
// Continuous relative position bias: t16s[m*6+h] = 16*sigmoid(cpb MLP).
// ---------------------------------------------------------------------------
__global__ __launch_bounds__(256) void k_cpb(const float* __restrict__ w1,
                                             const float* __restrict__ b1,
                                             const float* __restrict__ w2,
                                             float* __restrict__ t16s) {
  int g = blockIdx.x * 256 + threadIdx.x;
  if (g >= 961 * NHEAD) return;
  int m = g / NHEAD, h = g % NHEAD;
  int a = m / 31, b = m % 31;
  float va = (float)(a - 15) * (8.0f / 15.0f);
  float vb = (float)(b - 15) * (8.0f / 15.0f);
  float t0 = (va >= 0.f ? 1.f : -1.f) * log2f(fabsf(va) + 1.0f) * (1.0f / 3.0f);
  if (va == 0.f) t0 = 0.f;
  float t1 = (vb >= 0.f ? 1.f : -1.f) * log2f(fabsf(vb) + 1.0f) * (1.0f / 3.0f);
  if (vb == 0.f) t1 = 0.f;
  float acc = 0.f;
  for (int j = 0; j < 512; ++j) {
    float r = fmaxf(0.f, t0 * w1[2 * j] + t1 * w1[2 * j + 1] + b1[j]);
    acc += r * w2[h * 512 + j];
  }
  t16s[g] = 16.0f / (1.0f + __expf(-acc));
}

// ---------------------------------------------------------------------------
// Combined 16*sigmoid(rpb) + shift-mask table.
// Only 4 distinct window classes: cls = (wh==3)*2 + (ww==3).
// bmt[((cls*NH+h)*256+m)*256+n]
// ---------------------------------------------------------------------------
__global__ __launch_bounds__(256) void k_bmt(const float* __restrict__ t16s,
                                             float* __restrict__ bmt) {
  int g = blockIdx.x * 256 + threadIdx.x;   // 4*NH*256*256 threads
  int n = g & 255;
  int m = (g >> 8) & 255;
  int hc = g >> 16;
  int head = hc % NHEAD, cls = hc / NHEAD;
  int im = m >> 4, jm = m & 15, in_ = n >> 4, jn = n & 15;
  int idx = (im - in_ + 15) * 31 + (jm - jn + 15);
  float bias = t16s[idx * NHEAD + head];
  int eh = cls >> 1, ew = cls & 1;          // 1 if last window row/col
  int rm = (eh ? (im < 8 ? 1 : 2) : 0) * 3 + (ew ? (jm < 8 ? 1 : 2) : 0);
  int rn = (eh ? (in_ < 8 ? 1 : 2) : 0) * 3 + (ew ? (jn < 8 ? 1 : 2) : 0);
  bmt[g] = bias + ((rm != rn) ? -100.0f : 0.0f);
}

// ---------------------------------------------------------------------------
// Roll(-8,-8) + window partition + bf16 pack: xw[widx][n][c]
// ---------------------------------------------------------------------------
__global__ __launch_bounds__(256) void k_pack(const float* __restrict__ x,
                                              uint16_t* __restrict__ xw) {
  int g = blockIdx.x * 256 + threadIdx.x;      // TOKENS * 48 threads
  int tw = g / 48;
  int c4 = (g % 48) * 4;
  int widx = tw >> 8;
  int n = tw & 255;
  int b = widx >> 4;
  int wl = widx & 15;
  int wh = wl >> 2, wwj = wl & 3;
  int i = n >> 4, j = n & 15;
  int hs = (wh * 16 + i + 8) & 63;
  int ws_ = (wwj * 16 + j + 8) & 63;
  size_t src = ((size_t)((b * 64 + hs) * 64 + ws_)) * CH + c4;
  float4 vv = *(const float4*)(x + src);
  size_t dst = (size_t)tw * CH + c4;
  *(uint32_t*)(xw + dst)     = (uint32_t)f2bf(vv.x) | ((uint32_t)f2bf(vv.y) << 16);
  *(uint32_t*)(xw + dst + 2) = (uint32_t)f2bf(vv.z) | ((uint32_t)f2bf(vv.w) << 16);
}

// ---------------------------------------------------------------------------
// Generic bf16 GEMM:  C[M][N] = A[M][K] * W[N][K]^T (+bias) (+GELU).
// Block tile 128x64, 8 waves (4x2), wave tile 32x32, K chunks of 32 in LDS.
// LDS staging uses GLOBAL_LOAD_ASYNC_TO_LDS_B128 when available.
// ---------------------------------------------------------------------------
template <int ACT, int OUTBF>
__global__ __launch_bounds__(256) void k_gemm(const uint16_t* __restrict__ A,
                                              const uint16_t* __restrict__ Bw,
                                              const float* __restrict__ bias,
                                              float* __restrict__ Cf,
                                              uint16_t* __restrict__ Cb,
                                              int M, int N, int K) {
  __shared__ __align__(16) uint16_t sA[128 * 32];
  __shared__ __align__(16) uint16_t sB[64 * 32];
  const int tid = threadIdx.x;
  const int wave = tid >> 5, lane = tid & 31;
  const int half = lane >> 4, lr = lane & 15;
  const int wm = wave >> 1, wn = wave & 1;
  const int bm = blockIdx.x * 128, bn = blockIdx.y * 64;

  v8f acc[2][2];
  const v8f zv = {0.f, 0.f, 0.f, 0.f, 0.f, 0.f, 0.f, 0.f};
#pragma unroll
  for (int i = 0; i < 2; ++i)
#pragma unroll
    for (int j = 0; j < 2; ++j) acc[i][j] = zv;

  const int ar = tid >> 1, ac = (tid & 1) * 16;   // A: 16 elems/thread
  const int br = tid >> 2, bc = (tid & 3) * 8;    // B: 8 elems/thread

  for (int k0 = 0; k0 < K; k0 += 32) {
    const uint16_t* gA0 = A + (size_t)(bm + ar) * K + k0 + ac;
    const uint16_t* gB0 = Bw + (size_t)(bn + br) * K + k0 + bc;
#if USE_ASYNC_LDS
    __syncthreads();
    __builtin_amdgcn_global_load_async_to_lds_b128(
        (gv4i_p)gA0, (lv4i_p)(sA + ar * 32 + ac), 0, 0);
    __builtin_amdgcn_global_load_async_to_lds_b128(
        (gv4i_p)(gA0 + 8), (lv4i_p)(sA + ar * 32 + ac + 8), 0, 0);
    __builtin_amdgcn_global_load_async_to_lds_b128(
        (gv4i_p)gB0, (lv4i_p)(sB + br * 32 + bc), 0, 0);
#if __has_builtin(__builtin_amdgcn_s_wait_asynccnt)
    __builtin_amdgcn_s_wait_asynccnt(0);
#else
    asm volatile("s_wait_asynccnt 0" ::: "memory");
#endif
    __syncthreads();
#else
    uint4 a0 = *(const uint4*)gA0;
    uint4 a1 = *(const uint4*)(gA0 + 8);
    uint4 b0 = *(const uint4*)gB0;
    if (k0 + 32 < K) {
      __builtin_prefetch(gA0 + 32, 0, 1);
      __builtin_prefetch(gB0 + 32, 0, 1);
    }
    __syncthreads();
    *(uint4*)(sA + ar * 32 + ac)     = a0;
    *(uint4*)(sA + ar * 32 + ac + 8) = a1;
    *(uint4*)(sB + br * 32 + bc)     = b0;
    __syncthreads();
#endif

    FragU fa[2], fb[2];
#pragma unroll
    for (int i = 0; i < 2; ++i) {
      const int ms = wm * 32 + i * 16 + lr;
#pragma unroll
      for (int v = 0; v < 8; ++v)
        fa[i].u[v] = *(const uint32_t*)&sA[ms * 32 + kkA(v, half)];
    }
#pragma unroll
    for (int j = 0; j < 2; ++j) {
      const int ns = wn * 32 + j * 16 + lr;
#pragma unroll
      for (int v = 0; v < 8; ++v)
        fb[j].u[v] = *(const uint32_t*)&sB[ns * 32 + half * 16 + 2 * v];
    }
#pragma unroll
    for (int i = 0; i < 2; ++i)
#pragma unroll
      for (int j = 0; j < 2; ++j) acc[i][j] = wmma_bf16(fa[i], fb[j], acc[i][j]);
  }

#pragma unroll
  for (int i = 0; i < 2; ++i)
#pragma unroll
    for (int j = 0; j < 2; ++j)
#pragma unroll
      for (int r = 0; r < 8; ++r) {
        int m = bm + wm * 32 + i * 16 + r + half * 8;
        int n = bn + wn * 32 + j * 16 + lr;
        float x = acc[i][j][r] + (bias ? bias[n] : 0.0f);
        if (ACT == 1) x = 0.5f * x * (1.0f + erff(x * 0.70710678118f));
        if (OUTBF) Cb[(size_t)m * N + n] = f2bf(x);
        else       Cf[(size_t)m * N + n] = x;
      }
}

// ---------------------------------------------------------------------------
// Per-head L2 normalize (D=32) + bf16 pack. One wave per token:
// lane's k-th value (c = k*32+lane) belongs to head k, dim lane.
// ---------------------------------------------------------------------------
__global__ __launch_bounds__(256) void k_qknorm(const float* __restrict__ src,
                                                uint16_t* __restrict__ dst) {
  int wave = threadIdx.x >> 5, lane = threadIdx.x & 31;
  int t = blockIdx.x * 8 + wave;
  float v[NHEAD];
#pragma unroll
  for (int k = 0; k < NHEAD; ++k) v[k] = src[(size_t)t * CH + k * 32 + lane];
#pragma unroll
  for (int k = 0; k < NHEAD; ++k) {
    float ss = v[k] * v[k];
#pragma unroll
    for (int m = 1; m < 32; m <<= 1) ss += __shfl_xor(ss, m, 32);
    dst[(size_t)t * CH + k * 32 + lane] = f2bf(v[k] * rsqrtf(ss));
  }
}

// ---------------------------------------------------------------------------
// V transpose within each window: vt[widx][c][n] = bf16(v[widx][n][c]).
// ---------------------------------------------------------------------------
__global__ __launch_bounds__(256) void k_vtrans(const float* __restrict__ v,
                                                uint16_t* __restrict__ vt) {
  size_t g = (size_t)blockIdx.x * 256 + threadIdx.x;   // NWINTOT*CH*WINN
  int widx = (int)(g / (CH * WINN));
  int rem = (int)(g % (CH * WINN));
  int c = rem / WINN, n = rem % WINN;
  vt[g] = f2bf(v[(size_t)widx * WINN * CH + (size_t)n * CH + c]);
}

// ---------------------------------------------------------------------------
// Window attention. Grid = NWINTOT*NHEAD workgroups, 8 waves; each wave owns
// 32 queries, keys streamed in chunks of 32 (1 WMMA per 16x16 S tile, K=D=32).
//
// Bounded-logit softmax: cosine attention gives qn.kn in [-1,1], sigmoid bias
// in [0,16], and the diagonal entry equals scale + bias >= scale. So every
// row max lies in [scale, scale+16]; a FIXED shift M = scale+16 keeps all
// exponents within 16 of the true max -> exact softmax with NO row-max
// reduction and NO accumulator rescaling. Row sums reduced once at the end.
// ---------------------------------------------------------------------------
__global__ __launch_bounds__(256) void k_attn(const uint16_t* __restrict__ qn,
                                              const uint16_t* __restrict__ kn,
                                              const uint16_t* __restrict__ vt,
                                              const float* __restrict__ bmt,
                                              const float* __restrict__ lscale,
                                              uint16_t* __restrict__ xo) {
  const int wid = blockIdx.x;
  const int widx = wid / NHEAD, head = wid % NHEAD;
  const int wh = (widx & 15) >> 2, wwj = widx & 3;
  const int cls = ((wh == 3) ? 2 : 0) + ((wwj == 3) ? 1 : 0);
  const int wave = threadIdx.x >> 5, lane = threadIdx.x & 31;
  const int half = lane >> 4, lr = lane & 15;
  const int qbase = wave * 32;
  const float sc = __expf(fminf(lscale[head], 4.6051702f));   // exp(min(ls, ln100))
  const float Mshift = sc + 16.0f;

  __shared__ uint16_t pbuf[8][32][32];   // per-wave P staging (bf16)

  const size_t wofs = (size_t)widx * WINN * CH;
  const uint16_t* qg = qn + wofs + head * 32;
  const uint16_t* kg = kn + wofs + head * 32;
  const uint16_t* vg = vt + wofs + (size_t)head * 32 * WINN;
  const float* bmh = bmt + (size_t)(cls * NHEAD + head) * WINN * WINN;

  // resident Q A-fragments (2 row-blocks of 16)
  FragU aq[2];
#pragma unroll
  for (int rb = 0; rb < 2; ++rb) {
    const uint16_t* p = qg + (size_t)(qbase + rb * 16 + lr) * CH;
#pragma unroll
    for (int v = 0; v < 8; ++v)
      aq[rb].u[v] = *(const uint32_t*)(p + kkA(v, half));
  }

  const v8f zv = {0.f, 0.f, 0.f, 0.f, 0.f, 0.f, 0.f, 0.f};
  v8f oacc[2][2];
#pragma unroll
  for (int rb = 0; rb < 2; ++rb) { oacc[rb][0] = zv; oacc[rb][1] = zv; }
  float lsum[2][8];
#pragma unroll
  for (int rb = 0; rb < 2; ++rb)
#pragma unroll
    for (int r = 0; r < 8; ++r) lsum[rb][r] = 0.f;

  for (int kc = 0; kc < 8; ++kc) {
    const int k0 = kc * 32;
    // K B-fragments: B[k=d][n=key] from kn[key][d] (pairs over d contiguous)
    FragU bk[2];
#pragma unroll
    for (int t = 0; t < 2; ++t) {
      const uint16_t* p = kg + (size_t)(k0 + t * 16 + lr) * CH + half * 16;
#pragma unroll
      for (int v = 0; v < 8; ++v) bk[t].u[v] = *(const uint32_t*)(p + 2 * v);
    }

#pragma unroll
    for (int rb = 0; rb < 2; ++rb) {
      v8f s0 = wmma_bf16(aq[rb], bk[0], zv);
      v8f s1 = wmma_bf16(aq[rb], bk[1], zv);
      // bias+mask row pointer: row m = qbase+rb*16+half*8+r, col n = k0+t*16+lr
      const float* bmr = bmh + (size_t)(qbase + rb * 16 + half * 8) * WINN + k0 + lr;
#pragma unroll
      for (int r = 0; r < 8; ++r) {
        float p0 = __expf(s0[r] * sc + (bmr[r * WINN] - Mshift));
        float p1 = __expf(s1[r] * sc + (bmr[r * WINN + 16] - Mshift));
        lsum[rb][r] += p0 + p1;
        const int row = rb * 16 + r + half * 8;
        pbuf[wave][row][lr]      = f2bf(p0);
        pbuf[wave][row][16 + lr] = f2bf(p1);
      }
    }

    // reload P as A-fragments (per-wave LDS is in-order within the wave)
    FragU ap[2];
#pragma unroll
    for (int rb = 0; rb < 2; ++rb)
#pragma unroll
      for (int v = 0; v < 8; ++v)
        ap[rb].u[v] = *(const uint32_t*)&pbuf[wave][rb * 16 + lr][kkA(v, half)];

    // V B-fragments: B[k=key][n=d] from vt[d][key] (pairs over key contiguous)
    FragU bv[2];
#pragma unroll
    for (int dt = 0; dt < 2; ++dt) {
      const uint16_t* p = vg + (size_t)(dt * 16 + lr) * WINN + k0 + half * 16;
#pragma unroll
      for (int v = 0; v < 8; ++v) bv[dt].u[v] = *(const uint32_t*)(p + 2 * v);
    }
#pragma unroll
    for (int rb = 0; rb < 2; ++rb)
#pragma unroll
      for (int dt = 0; dt < 2; ++dt)
        oacc[rb][dt] = wmma_bf16(ap[rb], bv[dt], oacc[rb][dt]);
  }

  // single row-sum reduction (within each 16-lane group), normalize, store
#pragma unroll
  for (int rb = 0; rb < 2; ++rb)
#pragma unroll
    for (int r = 0; r < 8; ++r) {
#pragma unroll
      for (int msk = 1; msk < 16; msk <<= 1)
        lsum[rb][r] += __shfl_xor(lsum[rb][r], msk, 32);
    }
#pragma unroll
  for (int rb = 0; rb < 2; ++rb)
#pragma unroll
    for (int dt = 0; dt < 2; ++dt)
#pragma unroll
      for (int r = 0; r < 8; ++r) {
        const int m = qbase + rb * 16 + r + half * 8;
        const float val = oacc[rb][dt][r] / lsum[rb][r];
        xo[wofs + (size_t)m * CH + head * 32 + dt * 16 + lr] = f2bf(val);
      }
}

// ---------------------------------------------------------------------------
// Conditional LayerNorm residual: out = resid + (t*ww+wb)*norm(src) + (t*bw+bb)
// GATHER=1: src is in window layout, apply window-reverse + roll(+8,+8).
// ---------------------------------------------------------------------------
template <int GATHER>
__global__ __launch_bounds__(256) void k_cln(const float* __restrict__ src,
                                             const float* __restrict__ resid,
                                             const float* __restrict__ timev,
                                             const float* __restrict__ ww,
                                             const float* __restrict__ wb,
                                             const float* __restrict__ bw,
                                             const float* __restrict__ bb,
                                             float* __restrict__ out,
                                             uint16_t* __restrict__ out_bf) {
  int wave = threadIdx.x >> 5, lane = threadIdx.x & 31;
  int t = blockIdx.x * 8 + wave;
  int b = t >> 12;
  size_t soff;
  if (GATHER) {
    int hw = t & 4095;
    int hh = hw >> 6, wc = hw & 63;
    int hr = (hh + 56) & 63, wr = (wc + 56) & 63;
    int wh = hr >> 4, i = hr & 15, wwi = wr >> 4, j = wr & 15;
    int widx = b * 16 + wh * 4 + wwi;
    int n = i * 16 + j;
    soff = ((size_t)widx * WINN + n) * CH;
  } else {
    soff = (size_t)t * CH;
  }
  float vals[NHEAD];
  float s1 = 0.f, s2 = 0.f;
#pragma unroll
  for (int k = 0; k < NHEAD; ++k) {
    float x = src[soff + k * 32 + lane];
    vals[k] = x; s1 += x; s2 += x * x;
  }
#pragma unroll
  for (int m = 1; m < 32; m <<= 1) {
    s1 += __shfl_xor(s1, m, 32);
    s2 += __shfl_xor(s2, m, 32);
  }
  const float mean = s1 * (1.0f / CH);
  const float var = s2 * (1.0f / CH) - mean * mean;
  const float inv = rsqrtf(var + LN_EPS);
  const float tv = timev[b];
#pragma unroll
  for (int k = 0; k < NHEAD; ++k) {
    int c = k * 32 + lane;
    float xn = (vals[k] - mean) * inv;
    float o = resid[(size_t)t * CH + c] + (tv * ww[c] + wb[c]) * xn + (tv * bw[c] + bb[c]);
    out[(size_t)t * CH + c] = o;
    if (out_bf) out_bf[(size_t)t * CH + c] = f2bf(o);
  }
}

// ---------------------------------------------------------------------------
// Launcher
// ---------------------------------------------------------------------------
extern "C" void kernel_launch(void* const* d_in, const int* in_sizes, int n_in,
                              void* d_out, int out_size, void* d_ws, size_t ws_size,
                              hipStream_t stream) {
  const float* hidden = (const float*)d_in[0];
  const float* timev  = (const float*)d_in[1];
  const float* q_w    = (const float*)d_in[2];
  const float* q_b    = (const float*)d_in[3];
  const float* k_w    = (const float*)d_in[4];
  const float* v_w    = (const float*)d_in[5];
  const float* v_b    = (const float*)d_in[6];
  const float* lscale = (const float*)d_in[7];
  const float* cpb_w1 = (const float*)d_in[8];
  const float* cpb_b1 = (const float*)d_in[9];
  const float* cpb_w2 = (const float*)d_in[10];
  const float* proj_w = (const float*)d_in[11];
  const float* proj_b = (const float*)d_in[12];
  const float* ln1_ww = (const float*)d_in[13];
  const float* ln1_wb = (const float*)d_in[14];
  const float* ln1_bw = (const float*)d_in[15];
  const float* ln1_bb = (const float*)d_in[16];
  const float* fc1_w  = (const float*)d_in[17];
  const float* fc1_b  = (const float*)d_in[18];
  const float* fc2_w  = (const float*)d_in[19];
  const float* fc2_b  = (const float*)d_in[20];
  const float* ln2_ww = (const float*)d_in[21];
  const float* ln2_wb = (const float*)d_in[22];
  const float* ln2_bw = (const float*)d_in[23];
  const float* ln2_bb = (const float*)d_in[24];

  char* ws = (char*)d_ws;
  uint16_t* xw   = (uint16_t*)(ws + OFF_A);
  float*    qf   = (float*)(ws + OFF_B);
  float*    kf   = (float*)(ws + OFF_C);
  float*    vf   = (float*)(ws + OFF_D);
  uint16_t* qnb  = (uint16_t*)(ws + OFF_E);
  uint16_t* knb  = (uint16_t*)(ws + OFF_F);
  uint16_t* vtb  = (uint16_t*)(ws + OFF_G);
  float*    t16s = (float*)(ws + OFF_H);
  uint16_t* wbf  = (uint16_t*)(ws + OFF_W);
  float*    bmt  = (float*)(ws + OFF_I);
  uint16_t* xo   = (uint16_t*)(ws + OFF_B);   // reuse (qf dead after norm)
  float*    prjf = (float*)(ws + OFF_C);      // reuse (kf dead)
  float*    hbuf = (float*)(ws + OFF_D);      // reuse (vf dead)
  uint16_t* hbf  = (uint16_t*)(ws + OFF_A);   // reuse (xw dead)
  uint16_t* mlp  = (uint16_t*)(ws + OFF_B);   // spans B+C (xo, prjf dead)
  float*    fc2f = (float*)(ws + OFF_E);      // spans E+F (qnb, knb dead)
  float*    outp = (float*)d_out;

  // --- weights -> bf16 ---
  k_cvt<<<144, 256, 0, stream>>>(q_w, wbf + WOFF_Q, CH * CH);
  k_cvt<<<144, 256, 0, stream>>>(k_w, wbf + WOFF_K, CH * CH);
  k_cvt<<<144, 256, 0, stream>>>(v_w, wbf + WOFF_V, CH * CH);
  k_cvt<<<144, 256, 0, stream>>>(proj_w, wbf + WOFF_P, CH * CH);
  k_cvt<<<576, 256, 0, stream>>>(fc1_w, wbf + WOFF_F1, HIDCH * CH);
  k_cvt<<<576, 256, 0, stream>>>(fc2_w, wbf + WOFF_F2, CH * HIDCH);

  // --- CPB bias table, then combined bias+mask table ---
  k_cpb<<<23, 256, 0, stream>>>(cpb_w1, cpb_b1, cpb_w2, t16s);
  k_bmt<<<(4 * NHEAD * WINN * WINN) / 256, 256, 0, stream>>>(t16s, bmt);

  // --- roll + window partition + bf16 pack ---
  k_pack<<<TOKENS * 48 / 256, 256, 0, stream>>>(hidden, xw);

  // --- QKV GEMMs (f32 out, bias folded) ---
  dim3 gqkv(TOKENS / 128, CH / 64);
  k_gemm<0, 0><<<gqkv, 256, 0, stream>>>(xw, wbf + WOFF_Q, q_b, qf, nullptr,
                                         TOKENS, CH, CH);
  k_gemm<0, 0><<<gqkv, 256, 0, stream>>>(xw, wbf + WOFF_K, nullptr, kf, nullptr,
                                         TOKENS, CH, CH);
  k_gemm<0, 0><<<gqkv, 256, 0, stream>>>(xw, wbf + WOFF_V, v_b, vf, nullptr,
                                         TOKENS, CH, CH);

  // --- per-head L2 norm + bf16, V transpose ---
  k_qknorm<<<TOKENS / 8, 256, 0, stream>>>(qf, qnb);
  k_qknorm<<<TOKENS / 8, 256, 0, stream>>>(kf, knb);
  k_vtrans<<<(NWINTOT * CH * WINN) / 256, 256, 0, stream>>>(vf, vtb);

  // --- attention ---
  k_attn<<<NWINTOT * NHEAD, 256, 0, stream>>>(qnb, knb, vtb, bmt, lscale, xo);

  // --- output projection ---
  k_gemm<0, 0><<<gqkv, 256, 0, stream>>>(xo, wbf + WOFF_P, proj_b, prjf, nullptr,
                                         TOKENS, CH, CH);

  // --- cond-LN 1 (window-reverse gather) + residual -> h (f32 + bf16) ---
  k_cln<1><<<TOKENS / 8, 256, 0, stream>>>(prjf, hidden, timev, ln1_ww, ln1_wb,
                                           ln1_bw, ln1_bb, hbuf, hbf);

  // --- MLP: fc1 (GELU, bf16 out), fc2 (f32 out) ---
  dim3 gfc1(TOKENS / 128, HIDCH / 64);
  k_gemm<1, 1><<<gfc1, 256, 0, stream>>>(hbf, wbf + WOFF_F1, fc1_b, nullptr, mlp,
                                         TOKENS, HIDCH, CH);
  dim3 gfc2(TOKENS / 128, CH / 64);
  k_gemm<0, 0><<<gfc2, 256, 0, stream>>>(mlp, wbf + WOFF_F2, fc2_b, fc2f, nullptr,
                                         TOKENS, CH, HIDCH);

  // --- cond-LN 2 + residual -> d_out ---
  k_cln<0><<<TOKENS / 8, 256, 0, stream>>>(fc2f, hbuf, timev, ln2_ww, ln2_wb,
                                           ln2_bw, ln2_bb, outp, nullptr);
  (void)in_sizes; (void)n_in; (void)out_size; (void)ws_size;
}
